// WassersteinLoss_62637803045458
// MI455X (gfx1250) — compile-verified, matrix operand order checked by reading, not run on hardware
//
#include <hip/hip_runtime.h>
#include <hip/hip_bf16.h>

typedef __attribute__((ext_vector_type(16))) _Float16 v16h;
typedef __attribute__((ext_vector_type(8)))  float    v8f;

#define BB 32
#define NN 1024
#define DD 64
#define SINK_ITERS 100
#define INV_EPSILON 10.0f   // 1/0.1
#define EPS_DIV 1e-8f
#define EPS_NORM 1e-8f

// Raw hardware transcendentals (~1 ulp): values here are ~1e-4..1e0, far from
// denormal range, so the IEEE-safe expansions are pure overhead.
__device__ __forceinline__ float fast_sqrt(float x) { return __builtin_amdgcn_sqrtf(x); }
__device__ __forceinline__ float fast_rcp (float x) { return __builtin_amdgcn_rcpf(x); }

// ---------------------------------------------------------------------------
// Kernel A: per-row sums / squared norms + f16 conversion of raw inputs.
// grid (NN/256, BB, 2), block 256.  z=0 -> pred, z=1 -> target.
// ---------------------------------------------------------------------------
__global__ __launch_bounds__(256) void ws_rows_kernel(
    const float* __restrict__ pred, const float* __restrict__ targ,
    _Float16* __restrict__ x16, _Float16* __restrict__ y16,
    float* __restrict__ x2, float* __restrict__ y2,
    float* __restrict__ rsp, float* __restrict__ rsq) {
  const int which = blockIdx.z;
  const int b     = blockIdx.y;
  const int row   = blockIdx.x * blockDim.x + threadIdx.x;
  const float*  src   = which ? targ : pred;
  _Float16*     dst16 = which ? y16  : x16;
  float*        sqn   = which ? y2   : x2;
  float*        rs    = which ? rsq  : rsp;

  const size_t base = ((size_t)b * NN + row) * DD;
  const float4* p4 = (const float4*)(src + base);
  float s = 0.0f, s2 = 0.0f;
#pragma unroll
  for (int k = 0; k < DD / 4; ++k) {
    float4 v = p4[k];
    s  += v.x + v.y + v.z + v.w;
    s2 += v.x * v.x + v.y * v.y + v.z * v.z + v.w * v.w;
    dst16[base + 4 * k + 0] = (_Float16)v.x;
    dst16[base + 4 * k + 1] = (_Float16)v.y;
    dst16[base + 4 * k + 2] = (_Float16)v.z;
    dst16[base + 4 * k + 3] = (_Float16)v.w;
  }
  sqn[(size_t)b * NN + row] = s2;
  rs [(size_t)b * NN + row] = s;
}

// ---------------------------------------------------------------------------
// Kernel A2: reduce 1024 row-sums -> inv[b*2+which] = 1/(sum+eps).
// grid (BB, 2), block 256.  Fixed-order tree -> deterministic.
// (Scalar normalization: full-precision divide, done once per sample.)
// ---------------------------------------------------------------------------
__global__ __launch_bounds__(256) void ws_norm_kernel(
    const float* __restrict__ rsp, const float* __restrict__ rsq,
    float* __restrict__ inv) {
  const int which = blockIdx.y;
  const int b     = blockIdx.x;
  const float* rs = (which ? rsq : rsp) + (size_t)b * NN;
  __shared__ float sm[256];
  const int t = threadIdx.x;
  sm[t] = rs[t] + rs[t + 256] + rs[t + 512] + rs[t + 768];
  __syncthreads();
  for (int o = 128; o > 0; o >>= 1) {
    if (t < o) sm[t] += sm[t + o];
    __syncthreads();
  }
  if (t == 0) inv[b * 2 + which] = 1.0f / (sm[0] + EPS_NORM);
}

// ---------------------------------------------------------------------------
// Shared WMMA tile: 16x16 dot-product tile of x @ y^T over DD=64 (two k-steps).
// A layout (16-bit 16x32): lane L<16 -> row L, K {0..7,16..23}; L>=16 -> K+8.
// B (32x16) is the dual layout, so loading the y tile identically gives y^T.
// ---------------------------------------------------------------------------
__device__ __forceinline__ v8f wmma_tile_dot(const _Float16* __restrict__ xr,
                                             const _Float16* __restrict__ yr) {
  v8f acc = {};
#pragma unroll
  for (int k0 = 0; k0 < DD; k0 += 32) {
    v16h a, bf;
#pragma unroll
    for (int e = 0; e < 8; ++e) {
      a [e]     = xr[k0 + e];
      a [e + 8] = xr[k0 + 16 + e];
      bf[e]     = yr[k0 + e];
      bf[e + 8] = yr[k0 + 16 + e];
    }
    acc = __builtin_amdgcn_wmma_f32_16x16x32_f16(false, a, false, bf,
                                                 (short)0, acc, false, false);
  }
  return acc;
}

// ---------------------------------------------------------------------------
// Kernel B: build K[b][m][n] = exp(-C/eps), C = sqrt(max(a^2 x2 + b^2 y2 - 2ab dot, 0)).
// grid (NN/16, NN/128, BB), block 256 (8 waves, one 16x16 tile each).
// ---------------------------------------------------------------------------
__global__ __launch_bounds__(256) void ws_buildK_kernel(
    const _Float16* __restrict__ x16, const _Float16* __restrict__ y16,
    const float* __restrict__ x2, const float* __restrict__ y2,
    const float* __restrict__ inv, float* __restrict__ Kbuf) {
  const int lane = threadIdx.x & 31;
  const int wave = threadIdx.x >> 5;
  const int b  = blockIdx.z;
  const int tm = blockIdx.y * 8 + wave;
  const int tn = blockIdx.x;
  const int hl = lane & 15;
  const int kbase = (lane >> 4) * 8;

  const _Float16* xr = x16 + ((size_t)b * NN + tm * 16 + hl) * DD + kbase;
  const _Float16* yr = y16 + ((size_t)b * NN + tn * 16 + hl) * DD + kbase;
  v8f acc = wmma_tile_dot(xr, yr);

  const float ia = inv[b * 2 + 0], ib = inv[b * 2 + 1];
  const float ia2 = ia * ia, ib2 = ib * ib, iab2 = 2.0f * ia * ib;
  const float* x2b = x2 + (size_t)b * NN;
  const float* y2b = y2 + (size_t)b * NN;

  const int gn = tn * 16 + hl;
  const float yn2 = y2b[gn] * ib2;
  const int mbase = tm * 16 + ((lane >> 4) << 3);
  float* Kb = Kbuf + (size_t)b * NN * NN;
#pragma unroll
  for (int r = 0; r < 8; ++r) {
    const int gm = mbase + r;
    float sq = fmaf(x2b[gm], ia2, yn2) - iab2 * acc[r];
    float c  = fast_sqrt(fmaxf(sq, 0.0f));
    Kb[(size_t)gm * NN + gn] = __expf(-INV_EPSILON * c);
  }
}

// ---------------------------------------------------------------------------
// Kernel C: 100 Sinkhorn iterations. One 1024-thread block per sample; K is
// L2-resident (4 MB/sample, 128 MB total < 192 MB L2).  Phase1 (K^T u)
// column-coalesced; Phase2 (K v) per-lane float4 row walk (L0-line reuse).
// ---------------------------------------------------------------------------
__global__ __launch_bounds__(1024, 1) void ws_sinkhorn_kernel(
    const float* __restrict__ Kbuf,
    float* __restrict__ ubuf, float* __restrict__ vbuf) {
  __shared__ float su[NN];
  __shared__ float sv[NN];
  const int t = threadIdx.x;
  const int b = blockIdx.x;
  const float* Kb = Kbuf + (size_t)b * NN * NN;

  su[t] = 1.0f / (float)NN;
  __syncthreads();

  for (int it = 0; it < SINK_ITERS; ++it) {
    // v_t = 1 / (sum_i K[i][t] * u[i] + eps)   -- lanes span consecutive t
    float acc = 0.0f;
#pragma unroll 8
    for (int i = 0; i < NN; ++i)
      acc = fmaf(Kb[(size_t)i * NN + t], su[i], acc);
    sv[t] = fast_rcp(acc + EPS_DIV);
    __syncthreads();

    // u_t = 1 / (sum_j K[t][j] * v[j] + eps)   -- per-lane row walk (float4)
    const float4* Kr = (const float4*)(Kb + (size_t)t * NN);
    float acc2 = 0.0f;
#pragma unroll 4
    for (int j4 = 0; j4 < NN / 4; ++j4) {
      float4 kk = Kr[j4];
      acc2 = fmaf(kk.x, sv[4 * j4 + 0], acc2);
      acc2 = fmaf(kk.y, sv[4 * j4 + 1], acc2);
      acc2 = fmaf(kk.z, sv[4 * j4 + 2], acc2);
      acc2 = fmaf(kk.w, sv[4 * j4 + 3], acc2);
    }
    su[t] = fast_rcp(acc2 + EPS_DIV);
    __syncthreads();
  }
  ubuf[(size_t)b * NN + t] = su[t];
  vbuf[(size_t)b * NN + t] = sv[t];
}

// ---------------------------------------------------------------------------
// Kernel D: loss pass.  Recompute C tile via WMMA, load K, accumulate
// u_m * K_mn * v_n * C_mn.  Fixed-order wave+block reduction -> partials.
// grid (NN/16, NN/128, BB), block 256.
// ---------------------------------------------------------------------------
__global__ __launch_bounds__(256) void ws_loss_kernel(
    const _Float16* __restrict__ x16, const _Float16* __restrict__ y16,
    const float* __restrict__ x2, const float* __restrict__ y2,
    const float* __restrict__ inv, const float* __restrict__ Kbuf,
    const float* __restrict__ ubuf, const float* __restrict__ vbuf,
    float* __restrict__ partials) {
  const int lane = threadIdx.x & 31;
  const int wave = threadIdx.x >> 5;
  const int b  = blockIdx.z;
  const int tm = blockIdx.y * 8 + wave;
  const int tn = blockIdx.x;
  const int hl = lane & 15;
  const int kbase = (lane >> 4) * 8;

  const _Float16* xr = x16 + ((size_t)b * NN + tm * 16 + hl) * DD + kbase;
  const _Float16* yr = y16 + ((size_t)b * NN + tn * 16 + hl) * DD + kbase;
  v8f acc = wmma_tile_dot(xr, yr);

  const float ia = inv[b * 2 + 0], ib = inv[b * 2 + 1];
  const float ia2 = ia * ia, ib2 = ib * ib, iab2 = 2.0f * ia * ib;
  const float* x2b = x2 + (size_t)b * NN;
  const float* y2b = y2 + (size_t)b * NN;

  const int gn = tn * 16 + hl;
  const float yn2 = y2b[gn] * ib2;
  const float vn  = vbuf[(size_t)b * NN + gn];
  const int mbase = tm * 16 + ((lane >> 4) << 3);
  const float* Kb = Kbuf + (size_t)b * NN * NN;
  const float* ub = ubuf + (size_t)b * NN;

  float part = 0.0f;
#pragma unroll
  for (int r = 0; r < 8; ++r) {
    const int gm = mbase + r;
    float sq = fmaf(x2b[gm], ia2, yn2) - iab2 * acc[r];
    float c  = fast_sqrt(fmaxf(sq, 0.0f));
    float kv = Kb[(size_t)gm * NN + gn];
    part = fmaf(ub[gm] * kv * vn, c, part);
  }
  // fixed butterfly wave reduction (wave32)
  for (int off = 16; off > 0; off >>= 1)
    part += __shfl_xor(part, off, 32);

  __shared__ float wsum[8];
  if (lane == 0) wsum[wave] = part;
  __syncthreads();
  if (threadIdx.x == 0) {
    float s = 0.0f;
#pragma unroll
    for (int w = 0; w < 8; ++w) s += wsum[w];
    const int blin = (blockIdx.z * gridDim.y + blockIdx.y) * gridDim.x + blockIdx.x;
    partials[blin] = s;
  }
}

// ---------------------------------------------------------------------------
// Kernel E: deterministic final reduction of 16384 partials -> mean loss.
// 1 block, 1024 threads.
// ---------------------------------------------------------------------------
__global__ __launch_bounds__(1024) void ws_final_kernel(
    const float* __restrict__ partials, float* __restrict__ out) {
  __shared__ float sm[1024];
  const int t = threadIdx.x;
  float s = 0.0f;
#pragma unroll
  for (int k = 0; k < 16; ++k) s += partials[t + k * 1024];
  sm[t] = s;
  __syncthreads();
  for (int o = 512; o > 0; o >>= 1) {
    if (t < o) sm[t] += sm[t + o];
    __syncthreads();
  }
  if (t == 0) out[0] = sm[0] * (1.0f / (float)BB);
}

// ---------------------------------------------------------------------------
extern "C" void kernel_launch(void* const* d_in, const int* in_sizes, int n_in,
                              void* d_out, int out_size, void* d_ws, size_t ws_size,
                              hipStream_t stream) {
  (void)in_sizes; (void)n_in; (void)out_size; (void)ws_size;
  const float* pred = (const float*)d_in[0];
  const float* targ = (const float*)d_in[1];
  float* out = (float*)d_out;

  char* ws = (char*)d_ws;
  size_t off = 0;
  auto take = [&](size_t bytes) { size_t o = off; off += (bytes + 255) & ~(size_t)255; return o; };

  float*    Kbuf = (float*)   (ws + take((size_t)BB * NN * NN * 4));   // 128 MB
  _Float16* x16  = (_Float16*)(ws + take((size_t)BB * NN * DD * 2));   // 4 MB
  _Float16* y16  = (_Float16*)(ws + take((size_t)BB * NN * DD * 2));   // 4 MB
  float*    x2   = (float*)   (ws + take((size_t)BB * NN * 4));
  float*    y2   = (float*)   (ws + take((size_t)BB * NN * 4));
  float*    rsp  = (float*)   (ws + take((size_t)BB * NN * 4));
  float*    rsq  = (float*)   (ws + take((size_t)BB * NN * 4));
  float*    inv  = (float*)   (ws + take((size_t)BB * 2 * 4));
  float*    ubuf = (float*)   (ws + take((size_t)BB * NN * 4));
  float*    vbuf = (float*)   (ws + take((size_t)BB * NN * 4));
  float*    part = (float*)   (ws + take((size_t)16384 * 4));

  ws_rows_kernel<<<dim3(NN / 256, BB, 2), 256, 0, stream>>>(
      pred, targ, x16, y16, x2, y2, rsp, rsq);
  ws_norm_kernel<<<dim3(BB, 2), 256, 0, stream>>>(rsp, rsq, inv);
  ws_buildK_kernel<<<dim3(NN / 16, NN / 128, BB), 256, 0, stream>>>(
      x16, y16, x2, y2, inv, Kbuf);
  ws_sinkhorn_kernel<<<dim3(BB), 1024, 0, stream>>>(Kbuf, ubuf, vbuf);
  ws_loss_kernel<<<dim3(NN / 16, NN / 128, BB), 256, 0, stream>>>(
      x16, y16, x2, y2, inv, Kbuf, ubuf, vbuf, part);
  ws_final_kernel<<<1, 1024, 0, stream>>>(part, out);
}
